// CannyNet_76115410420178
// MI455X (gfx1250) — matrix operand order repaired and checked by reading, not compiled, not run on hardware
//
#include <hip/hip_runtime.h>

// ---------- types for WMMA fragments ----------
typedef __attribute__((ext_vector_type(16))) __bf16 v16bf;
typedef __attribute__((ext_vector_type(8)))  float  v8f;

struct alignas(16) U4b    { unsigned int x, y, z, w; };
struct alignas(32) Bits32 { U4b lo, hi; };            // raw bits of a 16xbf16 fragment
struct alignas(32) Pack32 { unsigned int w[8]; };     // 16 packed bf16

static __device__ __forceinline__ unsigned short f2bf(float f) {
  unsigned int u = __builtin_bit_cast(unsigned int, f);
  u += 0x7FFFu + ((u >> 16) & 1u);                    // round-to-nearest-even
  return (unsigned short)(u >> 16);
}

static __device__ __forceinline__ v16bf pack16(const unsigned short* s) {
  Pack32 p;
#pragma unroll
  for (int t = 0; t < 8; ++t)
    p.w[t] = (unsigned)s[2 * t] | ((unsigned)s[2 * t + 1] << 16);
  return __builtin_bit_cast(v16bf, p);
}

#define TS   64      // output tile
#define IMH  1024
#define IMW  1024
#define SU   112     // U stride (bf16): rows rel [-16,79], cols rel [-16,79]
#define SHT  104     // HB^T stride (bf16): [80 cols][96 rows], 208B = 13*16 (keeps b128 align)
#define SV   84      // VB stride (f32):  rows rel [-8,71],  cols rel [-8,71]
#define SM   68      // MAG stride (f32), rows rel [-1,64],  cols rel [-1,64]
#define LDS_BYTES (16 + 26880 + 16640 + 17952)   // flags + max(U,VB) + HB^T + MAG = 61488

__global__ __launch_bounds__(256) void canny_main(const float* __restrict__ img,
                                                  float* __restrict__ out,
                                                  unsigned int* __restrict__ gflags) {
  extern __shared__ char smem[];
  unsigned int*   sflags = (unsigned int*)smem;
  char*           regA   = smem + 16;
  unsigned short* Ubf    = (unsigned short*)regA;                 // live: stages 1-2
  float*          VB     = (float*)regA;                          // aliases U, live: 3-5
  unsigned short* HBt    = (unsigned short*)(smem + 16 + 26880);  // transposed, live: 2-3
  float*          MAG    = (float*)(smem + 16 + 26880 + 16640);   // live: 4-5

  const int tid  = threadIdx.x;
  const int lane = tid & 31;
  const int wave = tid >> 5;
  const int nlane = lane & 15;
  const int hi16  = lane >> 4;
  const int gr0 = blockIdx.y * TS;
  const int gc0 = blockIdx.x * TS;
  const long imgoff = (long)blockIdx.z * (long)(IMH * IMW);
  const float* im = img + imgoff;

  if (tid == 0) { sflags[0] = 0u; sflags[1] = 0u; }

  // ---- Stage 1: load u = img + 1 (OOB -> 0, matching zero-padded (x+1)*127.5 / 127.5)
  for (int p = tid; p < 96 * 96; p += 256) {
    int ru = p / 96, cu = p % 96;
    int gr = gr0 + ru - 16, gc = gc0 + cu - 16;
    float v = 0.0f;
    if ((unsigned)gr < (unsigned)IMH && (unsigned)gc < (unsigned)IMW)
      v = im[(long)gr * IMW + gc] + 1.0f;
    Ubf[ru * SU + cu] = f2bf(v);
  }

  // ---- constant banded-Gaussian fragments (per-lane, built once)
  const float g5[5] = {0.13533528f, 0.60653066f, 1.0f, 0.60653066f, 0.13533528f};
  v16bf bH, aV;
  {
    unsigned short s[16];
    const int kb0 = hi16 * 16;           // B layout: lanes 0-15 -> K 0..15, 16-31 -> K 16..31
#pragma unroll
    for (int i = 0; i < 16; ++i) {
      int j = kb0 + i - nlane - 6;       // B[k][n] = g[k - n - 6]
      s[i] = (j >= 0 && j <= 4) ? f2bf(g5[j]) : (unsigned short)0;
    }
    bH = pack16(s);
    const int kh = hi16 * 8;             // A layout: K halves {kh..kh+7, 16+kh..16+kh+7}
#pragma unroll
    for (int e = 0; e < 16; ++e) {
      int k = (e < 8) ? (kh + e) : (16 + kh + e - 8);
      int j = k - nlane - 6;             // A[m][k] = g[k - m - 6]
      s[e] = (j >= 0 && j <= 4) ? f2bf(g5[j]) : (unsigned short)0;
    }
    aV = pack16(s);
  }
  __syncthreads();

  // ---- Stage 2: horizontal 5-tap blur as WMMA, 6x5 16x16 tiles covering [-16,79]x[-8,71]
  for (int t = wave; t < 30; t += 8) {
    int r0 = -16 + (t / 5) * 16;
    int c0 = -8  + (t % 5) * 16;
    const int kh = hi16 * 8;
    int rowi = (r0 + nlane + 16) * SU;
    Bits32 ab;
    ab.lo = *(const U4b*)&Ubf[rowi + (c0 + 8  + 16 * hi16)];   // cols c0-8+16*hi16 .. +7
    ab.hi = *(const U4b*)&Ubf[rowi + (c0 + 16 + 16 * hi16)];   // cols c0+16*hi16   .. +7
    v16bf a = __builtin_bit_cast(v16bf, ab);
    v8f cz = {};
    v8f d = __builtin_amdgcn_wmma_f32_16x16x32_bf16(false, a, false, bH, (short)0, cz,
                                                    false, false);
    // store one packed column chunk (8 consecutive rows) per lane, branchless OOB->0
    int cc = c0 + nlane;
    int rbase = r0 + 8 * hi16;                          // rows rbase..rbase+7
    bool colok = (unsigned)(gc0 + cc) < (unsigned)IMW;
    unsigned short sv[8];
#pragma unroll
    for (int v = 0; v < 8; ++v) {
      bool ok = colok && ((unsigned)(gr0 + rbase + v) < (unsigned)IMH);
      sv[v] = f2bf(ok ? d[v] : 0.0f);
    }
    U4b pk;
    pk.x = (unsigned)sv[0] | ((unsigned)sv[1] << 16);
    pk.y = (unsigned)sv[2] | ((unsigned)sv[3] << 16);
    pk.z = (unsigned)sv[4] | ((unsigned)sv[5] << 16);
    pk.w = (unsigned)sv[6] | ((unsigned)sv[7] << 16);
    *(U4b*)&HBt[(cc + 8) * SHT + (rbase + 16)] = pk;
  }
  __syncthreads();

  // ---- Stage 3: vertical blur as WMMA, 5x5 tiles covering [-8,71]^2 (writes alias U)
  for (int t = wave; t < 25; t += 8) {
    int r0 = -8 + (t / 5) * 16;
    int c0 = -8 + (t % 5) * 16;
    const int kb0 = hi16 * 16;
    const unsigned short* colp = &HBt[(c0 + nlane + 8) * SHT];
    Bits32 bb;                                          // B[k][n]=hb[r0-8+k][c0+n], K={8h..8h+7, 16+8h..16+8h+7}
    bb.lo = *(const U4b*)&colp[r0 + 8 + 8 * hi16];
    bb.hi = *(const U4b*)&colp[r0 + 8 + 16 + 8 * hi16];
    v16bf b = __builtin_bit_cast(v16bf, bb);
    v8f cz = {};
    v8f d = __builtin_amdgcn_wmma_f32_16x16x32_bf16(false, aV, false, b, (short)0, cz,
                                                    false, false);
    int cc = c0 + nlane;
    bool colok = (unsigned)(gc0 + cc) < (unsigned)IMW;
#pragma unroll
    for (int v = 0; v < 8; ++v) {
      int rr = r0 + v + 8 * hi16;
      bool ok = colok && ((unsigned)(gr0 + rr) < (unsigned)IMH);
      VB[(rr + 8) * SV + (cc + 8)] = ok ? d[v] : 0.0f;
    }
  }
  __syncthreads();

  // ---- Stage 4: Sobel + gradient magnitude on [-1,64]^2
  for (int p = tid; p < 66 * 66; p += 256) {
    int r = p / 66 - 1, c = p % 66 - 1;
    const float* vb = &VB[(r + 8) * SV + (c + 8)];
    float a00 = vb[-SV - 1], a01 = vb[-SV], a02 = vb[-SV + 1];
    float a10 = vb[-1],                     a12 = vb[1];
    float a20 = vb[SV - 1],  a21 = vb[SV],  a22 = vb[SV + 1];
    float gx = (a00 + 2.f * a10 + a20) - (a02 + 2.f * a12 + a22);
    float gy = (a00 + 2.f * a01 + a02) - (a20 + 2.f * a21 + a22);
    float m = sqrtf(gx * gx + gy * gy);
    bool ok = ((unsigned)(gr0 + r) < (unsigned)IMH) && ((unsigned)(gc0 + c) < (unsigned)IMW);
    MAG[(r + 1) * SM + (c + 1)] = ok ? m : 0.0f;
  }
  __syncthreads();

  // ---- Stage 5: NMS + threshold (binary out), collect min/max flags
  // byte-packed +-(dr*SM+dc) table, n=0..7 -> {1,69,68,67,-1,-69,-68,-67}
  const unsigned long long TBL = 0xBDBCBBFF43444501ull;
  bool has1 = false, has0 = false;
  for (int p = tid; p < TS * TS; p += 256) {
    int r = p >> 6, c = p & 63;
    int base = (r + 1) * SM + (c + 1);
    float m = MAG[base];
    float o = 0.0f;
    if (m * 127.5f >= 400.0f) {                       // thr>0 requires thin>=400
      const float* vb = &VB[(r + 8) * SV + (c + 8)];
      float a00 = vb[-SV - 1], a01 = vb[-SV], a02 = vb[-SV + 1];
      float a10 = vb[-1],                     a12 = vb[1];
      float a20 = vb[SV - 1],  a21 = vb[SV],  a22 = vb[SV + 1];
      float gx = (a00 + 2.f * a10 + a20) - (a02 + 2.f * a12 + a22);
      float gy = (a00 + 2.f * a01 + a02) - (a20 + 2.f * a21 + a22);
      // octant n = round((atan2(gy,gx)*180/3.14159+180)/45) mod 8, via comparisons
      float ax = fabsf(gx), ay = fabsf(gy);
      const float T = 0.41421356f;                    // tan(22.5 deg)
      int n;
      if (ay <= ax * T)       n = (gx >= 0.f) ? 4 : 0;
      else if (ax <= ay * T)  n = (gy >  0.f) ? 6 : 2;
      else if (gx > 0.f)      n = (gy >  0.f) ? 5 : 3;
      else                    n = (gy >  0.f) ? 7 : 1;
      int didx = (int)(signed char)((TBL >> (8 * n)) & 0xFF);
      float pos = m - MAG[base + didx];
      float neg = m - MAG[base - didx];
      if (fminf(pos, neg) > 0.0f) o = 1.0f;
    }
    out[imgoff + (long)(gr0 + r) * IMW + (gc0 + c)] = o;
    has1 |= (o != 0.0f);
    has0 |= (o == 0.0f);
  }
  if (has1) atomicOr(&sflags[0], 1u);
  if (has0) atomicOr(&sflags[1], 1u);
  __syncthreads();
  if (tid == 0) {
    if (sflags[0]) atomicOr(&gflags[0], 1u);
    if (sflags[1]) atomicOr(&gflags[1], 1u);
  }
}

__global__ void canny_zero(unsigned int* gflags) {
  if (threadIdx.x < 2) gflags[threadIdx.x] = 0u;
}

__global__ void canny_norm4(float4* __restrict__ out,
                            const unsigned int* __restrict__ gflags, int n4) {
  int i = blockIdx.x * blockDim.x + threadIdx.x;
  if (i >= n4) return;
  float mx = gflags[0] ? 1.0f : 0.0f;
  float mn = gflags[1] ? 0.0f : 1.0f;
  float inv_is_nan_safe = mx - mn;                    // 0 -> 0/0 = NaN, matching reference
  float4 v = out[i];
  v.x = (v.x - mn) / inv_is_nan_safe;
  v.y = (v.y - mn) / inv_is_nan_safe;
  v.z = (v.z - mn) / inv_is_nan_safe;
  v.w = (v.w - mn) / inv_is_nan_safe;
  out[i] = v;
}

extern "C" void kernel_launch(void* const* d_in, const int* in_sizes, int n_in,
                              void* d_out, int out_size, void* d_ws, size_t ws_size,
                              hipStream_t stream) {
  const float* img = (const float*)d_in[0];           // [4,3,1024,1024] f32
  float* out = (float*)d_out;
  unsigned int* gflags = (unsigned int*)d_ws;

  canny_zero<<<1, 32, 0, stream>>>(gflags);
  dim3 grid(IMW / TS, IMH / TS, 12);                  // 16 x 16 tiles x (B*C)
  canny_main<<<grid, 256, LDS_BYTES, stream>>>(img, out, gflags);
  int n4 = out_size / 4;                              // 12,582,912 is divisible by 4
  canny_norm4<<<(n4 + 255) / 256, 256, 0, stream>>>((float4*)out, gflags, n4);
}
